// KA_GNN_two_69080253989644
// MI455X (gfx1250) — matrix operand order repaired and checked by reading, not compiled, not run on hardware
//
#include <hip/hip_runtime.h>
#include <hip/hip_bf16.h>

typedef __attribute__((ext_vector_type(16))) _Float16 v16h;
typedef __attribute__((ext_vector_type(8)))  _Float16 v8h;
typedef __attribute__((ext_vector_type(8)))  float    v8f;

// Problem constants
#define BATCH 128
#define NNODE 256
#define NCH   4
#define INDIM 64
#define HDIM  128
#define GRID  8

#define XS_STRIDE 129  // x-tile channel stride (conflict-free staging stores)
#define FT_STRIDE 136  // f16 transpose-staging row stride (16B aligned rows)
#define AT_STRIDE 40   // 16x32 A-tile rows padded (bank spread, 16B aligned)

// B-fragment K mapping (16-bit 32x16): lanes 0-15 hold K=0..15 (elements in
// order), lanes 16-31 hold K=16..31.
__device__ __forceinline__ int kslotB(int e, int laneHi) {
    return e + (laneHi ? 16 : 0);
}

// ---------------------------------------------------------------------------
// Pack W[2, HDIM, in_dim, GRID] (fp32) into f16 WMMA B-fragment layout:
// dst[((kc*8 + colTile)*32 + lane)*16 + e] = Wf[K][o]
//   K = kc*32 + kslotB(e, lane>=16),  o = colTile*16 + (lane&15)
//   Wf[K][o]: half = K/(in_dim*8); i = (K/8)%in_dim; k = K%8
// ---------------------------------------------------------------------------
__global__ void pack_weights_kernel(const float* __restrict__ W,
                                    _Float16* __restrict__ dst,
                                    int in_dim, int ktot) {
    int idx = blockIdx.x * blockDim.x + threadIdx.x;
    int total = ktot * HDIM;
    if (idx >= total) return;
    int e    = idx & 15;
    int lane = (idx >> 4) & 31;
    int ot   = (idx >> 9) & 7;
    int kc   = idx >> 12;
    int K    = kc * 32 + kslotB(e, lane >= 16);
    int o    = ot * 16 + (lane & 15);
    int span = in_dim * GRID;
    int half = K / span;
    int r    = K % span;
    int i    = r >> 3;       // / GRID
    int k    = r & 7;        // % GRID
    float w  = W[(((size_t)half * HDIM + o) * in_dim + i) * GRID + k];
    dst[idx] = (_Float16)w;
}

// ---------------------------------------------------------------------------
// Fourier-feature GEMM:  out[row, o] = sum_K feat(x[row], K) * Wf[K][o]
// Block = 256 threads = 8 waves, block owns 128 rows x 128 cols.
// Each WAVE owns a distinct 16-row tile and iterates all 8 col tiles with 8
// accumulators -> each sin/cos feature is computed exactly ONCE and feeds 16
// WMMAs per chunk pair (cos chunk kc and sin chunk kc+nchunks_half share
// angles: compute angle once, take v_cos + v_sin; branch-free).
// WF32: write out_f32[row][o] (node-major, for residual reads).
// WF16: write out_f16 TRANSPOSED as f_T[b][o][n] (n contiguous) so the
//       aggregation B fragment is one aligned 32B global load.
// ---------------------------------------------------------------------------
template <bool WF32, bool WF16>
__global__ void fourier_gemm_kernel(const float* __restrict__ x,
                                    const _Float16* __restrict__ pw,
                                    float* __restrict__ out_f32,
                                    _Float16* __restrict__ out_f16,
                                    int in_shift, int nchunks_half) {
    extern __shared__ float xs[];                 // [in_dim][XS_STRIDE]
    const int in_dim  = 1 << in_shift;
    const int rowbase = blockIdx.x * 128;
    const int t = threadIdx.x;

    for (int idx = t; idx < 128 * in_dim; idx += 256) {
        int r = idx >> in_shift;
        int i = idx & (in_dim - 1);
        xs[i * XS_STRIDE + r] = x[(size_t)rowbase * in_dim + idx];
    }
    __syncthreads();

    const int w      = t >> 5;
    const int lane   = t & 31;
    const int laneHi = lane >> 4;                 // 0 or 1
    const int row    = lane & 15;
    const int wrow   = w * 16 + row;              // this lane's row in block
    const size_t halfOff = (size_t)nchunks_half * 8 * 32 * 16;  // sin weights

    v8f zero = {};
    v8f acc[8];
    #pragma unroll
    for (int ot = 0; ot < 8; ++ot) acc[ot] = zero;

    for (int kc = 0; kc < nchunks_half; ++kc) {
        const int i0 = kc * 4 + laneHi * 2;       // channel for elements 0..7
        const float x0 = xs[i0 * XS_STRIDE + wrow];
        const float x1 = xs[(i0 + 1) * XS_STRIDE + wrow];  // elements 8..15
        v16h ac, as;
        #pragma unroll
        for (int e = 0; e < 8; ++e) {
            float v0 = x0 * (float)(e + 1);
            float v1 = x1 * (float)(e + 1);
            ac[e]     = (_Float16)__cosf(v0);
            as[e]     = (_Float16)__sinf(v0);
            ac[e + 8] = (_Float16)__cosf(v1);
            as[e + 8] = (_Float16)__sinf(v1);
        }
        const _Float16* pk = pw + ((size_t)kc * 8 * 32 + lane) * 16;
        #pragma unroll
        for (int ot = 0; ot < 8; ++ot) {
            v16h bc = *(const v16h*)(pk + (size_t)ot * 512);
            v16h bs = *(const v16h*)(pk + (size_t)ot * 512 + halfOff);
            acc[ot] = __builtin_amdgcn_wmma_f32_16x16x32_f16(
                false, ac, false, bc, (short)0, acc[ot], false, false);
            acc[ot] = __builtin_amdgcn_wmma_f32_16x16x32_f16(
                false, as, false, bs, (short)0, acc[ot], false, false);
        }
    }

    if (WF32) {
        #pragma unroll
        for (int ot = 0; ot < 8; ++ot) {
            const int col = ot * 16 + row;
            #pragma unroll
            for (int d = 0; d < 8; ++d) {
                size_t idx = ((size_t)rowbase + w * 16 + d + laneHi * 8) * HDIM + col;
                out_f32[idx] = acc[ot][d];
            }
        }
    }
    if (WF16) {
        // transpose 128x128 f16 tile through LDS (reuse xs buffer)
        __syncthreads();                       // all waves done with xs
        _Float16* ft = (_Float16*)xs;          // [128 o][FT_STRIDE]
        #pragma unroll
        for (int ot = 0; ot < 8; ++ot) {
            const int col = ot * 16 + row;
            #pragma unroll
            for (int d = 0; d < 8; ++d)
                ft[col * FT_STRIDE + w * 16 + d + laneHi * 8] = (_Float16)acc[ot][d];
        }
        __syncthreads();
        // coalesced write-out: f_T[(b*128 + o)*256 + nlocal]
        const int bglob = rowbase >> 8;        // batch of this block
        const int nloc0 = rowbase & 255;       // node offset of this block
        const int o  = t >> 1;
        const int nh = (t & 1) * 64;
        _Float16* gp = out_f16 + ((size_t)bglob * HDIM + o) * NNODE + nloc0 + nh;
        #pragma unroll
        for (int j = 0; j < 4; ++j)
            *(v16h*)(gp + j * 16) = *(const v16h*)(&ft[o * FT_STRIDE + nh + j * 16]);
    }
}

// ---------------------------------------------------------------------------
// Graph aggregation + residual + leaky ReLU:
//   m[b,mm,o] = sum_{n,c} A[b,n,c,mm] * f[b,n,o];  h_out = leaky(m + h_in)
// Block = (batch b, m-tile of 16 rows) x all 128 cols -> A streamed ONCE
// (268 MB over both layers = the 23.3 TB/s bandwidth floor of the problem).
// AT is double-buffered -> single barrier per chunk (a wave can be at most
// one barrier ahead, so it only ever writes the buffer nobody is reading).
// A fragment: two aligned 16B LDS loads from padded AT tile (c-reduced via
// float4 b128 loads along the contiguous m axis).
// B fragment: one aligned 32B global load from transposed f_T (L2-resident).
// ---------------------------------------------------------------------------
__global__ void aggregate_kernel(const float* __restrict__ A,
                                 const _Float16* __restrict__ fT,
                                 const float* __restrict__ h_in,
                                 float* __restrict__ h_out) {
    __shared__ __attribute__((aligned(32))) _Float16 AT[2][16 * AT_STRIDE];

    const int b     = blockIdx.x >> 4;
    const int mbase = (blockIdx.x & 15) * 16;
    const int t     = threadIdx.x;
    const int w     = t >> 5;
    const int lane  = t & 31;
    const int laneHi = lane >> 4;
    const int row   = lane & 15;
    const int col   = w * 16 + row;

    const _Float16* fcol = fT + ((size_t)b * HDIM + col) * NNODE + laneHi * 16;

    v8f acc = {};
    #pragma unroll
    for (int nc = 0; nc < 8; ++nc) {
        const int nbase = nc * 32;
        _Float16* ATb = AT[nc & 1];

        // c-reduction of A[b, nbase..+32, 0..3, mbase..+16] -> AT[m][n]
        // threads 0..127: one (n, m-quad) each; b128 loads along contiguous m
        if (t < 128) {
            int n  = t >> 2;          // 0..31
            int mq = (t & 3) * 4;     // m quad base
            const float* ap = A + (((size_t)b * NNODE + (nbase + n)) * NCH) * NNODE
                                + (mbase + mq);
            __builtin_prefetch(ap + (size_t)32 * NCH * NNODE, 0, 1);
            float4 c0 = *(const float4*)(ap);
            float4 c1 = *(const float4*)(ap + NNODE);
            float4 c2 = *(const float4*)(ap + 2 * NNODE);
            float4 c3 = *(const float4*)(ap + 3 * NNODE);
            ATb[(mq + 0) * AT_STRIDE + n] = (_Float16)(c0.x + c1.x + c2.x + c3.x);
            ATb[(mq + 1) * AT_STRIDE + n] = (_Float16)(c0.y + c1.y + c2.y + c3.y);
            ATb[(mq + 2) * AT_STRIDE + n] = (_Float16)(c0.z + c1.z + c2.z + c3.z);
            ATb[(mq + 3) * AT_STRIDE + n] = (_Float16)(c0.w + c1.w + c2.w + c3.w);
        }
        // B fragment direct from global (L2-hot): contiguous K per lane half
        v16h bb = *(const v16h*)(fcol + nbase);
        __syncthreads();

        // A fragment: elements 0..7 = AT[row][laneHi*16 + 0..7],
        //             elements 8..15 = AT[row][laneHi*16 + 8..15]
        const _Float16* abase = &ATb[row * AT_STRIDE + laneHi * 16];
        v8h alo = *(const v8h*)(abase);
        v8h ahi = *(const v8h*)(abase + 8);
        v16h a = __builtin_shufflevector(alo, ahi,
                    0, 1, 2, 3, 4, 5, 6, 7, 8, 9, 10, 11, 12, 13, 14, 15);

        acc = __builtin_amdgcn_wmma_f32_16x16x32_f16(
            false, a, false, bb, (short)0, acc, false, false);
    }

    #pragma unroll
    for (int d = 0; d < 8; ++d) {
        int m = mbase + d + laneHi * 8;
        size_t idx = ((size_t)b * NNODE + m) * HDIM + col;
        float v = acc[d] + h_in[idx];
        h_out[idx] = v > 0.f ? v : 0.01f * v;
    }
}

// ---------------------------------------------------------------------------
// Masked mean-pool over first mol_size[b] nodes + g=1 KAN head + sigmoid.
// ---------------------------------------------------------------------------
__global__ void pool_head_kernel(const float* __restrict__ h,
                                 const int* __restrict__ mol_size,
                                 const float* __restrict__ W_out,
                                 const float* __restrict__ b_out,
                                 float* __restrict__ out) {
    const int b = blockIdx.x;
    const int o = threadIdx.x;          // 0..127
    const int mol = mol_size[b];
    const float* hp = h + (size_t)b * NNODE * HDIM + o;
    float s = 0.f;
    for (int n = 0; n < mol; ++n) s += hp[(size_t)n * HDIM];
    float y = s / (float)mol;
    // W_out: [2, 1, HDIM, 1]
    float c = __cosf(y) * W_out[o] + __sinf(y) * W_out[HDIM + o];

    __shared__ float red[HDIM];
    red[o] = c;
    __syncthreads();
    for (int st = 64; st > 0; st >>= 1) {
        if (o < st) red[o] += red[o + st];
        __syncthreads();
    }
    if (o == 0) {
        float z = red[0] + b_out[0];
        out[b] = 1.f / (1.f + __expf(-z));
    }
}

// ---------------------------------------------------------------------------
extern "C" void kernel_launch(void* const* d_in, const int* in_sizes, int n_in,
                              void* d_out, int out_size, void* d_ws, size_t ws_size,
                              hipStream_t stream) {
    const float* V     = (const float*)d_in[0];
    const float* A     = (const float*)d_in[1];
    const int*   mol   = (const int*)d_in[2];
    const float* W_in  = (const float*)d_in[3];
    const float* W_g   = (const float*)d_in[4];   // [2, 2, H, H, G]
    const float* W_out = (const float*)d_in[5];
    const float* b_out = (const float*)d_in[6];
    float* out = (float*)d_out;

    // workspace layout
    char* ws = (char*)d_ws;
    const size_t H_BYTES  = (size_t)BATCH * NNODE * HDIM * sizeof(float);   // 16 MiB
    const size_t F_BYTES  = (size_t)BATCH * NNODE * HDIM * sizeof(_Float16);// 8 MiB
    float*    hA    = (float*)ws;
    float*    hB    = (float*)(ws + H_BYTES);
    _Float16* fbuf  = (_Float16*)(ws + 2 * H_BYTES);          // f_T[b][o][n]
    _Float16* pw_in = (_Float16*)(ws + 2 * H_BYTES + F_BYTES);
    const int KIN = 2 * INDIM * GRID;   // 1024
    const int KG  = 2 * HDIM  * GRID;   // 2048
    _Float16* pw_g0 = pw_in + (size_t)KIN * HDIM;
    _Float16* pw_g1 = pw_g0 + (size_t)KG * HDIM;
    const size_t WG_LAYER_STRIDE = (size_t)2 * HDIM * HDIM * GRID;  // floats

    const int ROW_BLOCKS = (BATCH * NNODE) / 128;   // 256

    // 1) pack weights into WMMA B-fragment layout
    pack_weights_kernel<<<(KIN * HDIM + 255) / 256, 256, 0, stream>>>(
        W_in, pw_in, INDIM, KIN);
    pack_weights_kernel<<<(KG * HDIM + 255) / 256, 256, 0, stream>>>(
        W_g, pw_g0, HDIM, KG);
    pack_weights_kernel<<<(KG * HDIM + 255) / 256, 256, 0, stream>>>(
        W_g + WG_LAYER_STRIDE, pw_g1, HDIM, KG);

    // 2) input layer:  hA = kan_linear(V, W_in)
    fourier_gemm_kernel<true, false>
        <<<ROW_BLOCKS, 256, INDIM * XS_STRIDE * sizeof(float), stream>>>(
        V, pw_in, hA, nullptr, 6, KIN / 64);

    // 3) GNN layer 0:  f_T = fourier(hA, Wg0);  hB = leaky(A@f + hA)
    fourier_gemm_kernel<false, true>
        <<<ROW_BLOCKS, 256, HDIM * XS_STRIDE * sizeof(float), stream>>>(
        hA, pw_g0, nullptr, fbuf, 7, KG / 64);
    aggregate_kernel<<<BATCH * 16, 256, 0, stream>>>(A, fbuf, hA, hB);

    // 4) GNN layer 1:  f_T = fourier(hB, Wg1);  hA = leaky(A@f + hB)
    fourier_gemm_kernel<false, true>
        <<<ROW_BLOCKS, 256, HDIM * XS_STRIDE * sizeof(float), stream>>>(
        hB, pw_g1, nullptr, fbuf, 7, KG / 64);
    aggregate_kernel<<<BATCH * 16, 256, 0, stream>>>(A, fbuf, hB, hA);

    // 5) pooling + output head
    pool_head_kernel<<<BATCH, HDIM, 0, stream>>>(hA, mol, W_out, b_out, out);
}
